// DYDConv2d_71476845740317
// MI455X (gfx1250) — matrix operand order repaired and checked by reading, not compiled, not run on hardware
//
#include <hip/hip_runtime.h>
#include <hip/hip_bf16.h>

typedef __attribute__((ext_vector_type(16))) __bf16 v16bf;
typedef __attribute__((ext_vector_type(8)))  float  v8f;

#define B_   16
#define C_   256
#define H_   64
#define W_   64
#define HW_  4096
#define O_   256
#define KF   4
#define HID  65

union BF32 { uint4 q[2]; v16bf v; };

__device__ __forceinline__ unsigned short f32_to_bf16(float f) {
    unsigned int u = __float_as_uint(f);
    u += 0x7FFFu + ((u >> 16) & 1u);          // round-to-nearest-even
    return (unsigned short)(u >> 16);
}

// ---------------- global average pool: x[b][c][hw] -> pooled[b][c] ----------
__global__ void pool_kernel(const float* __restrict__ x, float* __restrict__ pooled) {
    const int bc = blockIdx.x;                 // b*C_ + c
    const float* p = x + (size_t)bc * HW_;
    float s = 0.f;
    for (int i = threadIdx.x; i < HW_; i += 256) s += p[i];
    __shared__ float red[256];
    red[threadIdx.x] = s;
    __syncthreads();
    for (int off = 128; off > 0; off >>= 1) {
        if (threadIdx.x < off) red[threadIdx.x] += red[threadIdx.x + off];
        __syncthreads();
    }
    if (threadIdx.x == 0) pooled[bc] = red[0] * (1.0f / HW_);
}

// ------------- transpose + bf16 convert: x[b][c][hw] -> xT[b][hw][c] --------
__global__ void transpose_kernel(const float* __restrict__ x,
                                 unsigned short* __restrict__ xT) {
    const int b   = blockIdx.z;
    const int c0  = blockIdx.y * 64;
    const int hw0 = blockIdx.x * 64;
    __shared__ unsigned short tile[64 * 65];   // [c_local][hw_local] padded
    const float* xb = x + ((size_t)b * C_ + c0) * HW_ + hw0;
    for (int i = 0; i < 16; ++i) {
        int idx = threadIdx.x + i * 256;
        int cl = idx >> 6, hl = idx & 63;
        tile[cl * 65 + hl] = f32_to_bf16(xb[(size_t)cl * HW_ + hl]);
    }
    __syncthreads();
    const int hl = threadIdx.x >> 2;
    const int cc = (threadIdx.x & 3) * 16;
    unsigned short vals[16] __attribute__((aligned(16)));
    for (int j = 0; j < 16; ++j) vals[j] = tile[(cc + j) * 65 + hl];
    uint4* dst = (uint4*)(xT + ((size_t)b * HW_ + hw0 + hl) * C_ + c0 + cc);
    dst[0] = *(const uint4*)&vals[0];
    dst[1] = *(const uint4*)&vals[8];
}

// ------------- SE MLP + softmax: pooled -> prob[b][k][o] --------------------
__global__ void prob_kernel(const float* __restrict__ pooled,
                            const float* __restrict__ fc1_w,
                            const float* __restrict__ fc2_w,
                            const float* __restrict__ fc2_b,
                            float* __restrict__ prob) {
    const int b = blockIdx.x;
    const int t = threadIdx.x;
    __shared__ float sp[C_];
    __shared__ float sh[HID];
    sp[t] = pooled[b * C_ + t];
    __syncthreads();
    if (t < HID) {
        float a = 0.f;
        const float* w = fc1_w + t * C_;
        for (int c = 0; c < C_; ++c) a += sp[c] * w[c];
        sh[t] = fmaxf(a, 0.f);
    }
    __syncthreads();
    float y[KF];
    for (int k = 0; k < KF; ++k) {
        const int i = k * O_ + t;              // y[b, k, o=t]
        float a = fc2_b[i];
        const float* w = fc2_w + i * HID;
        for (int j = 0; j < HID; ++j) a += sh[j] * w[j];
        y[k] = a * (1.0f / 30.0f);             // / TEMPERATURE
    }
    float m = y[0];
    for (int k = 1; k < KF; ++k) m = fmaxf(m, y[k]);
    float e[KF], s = 0.f;
    for (int k = 0; k < KF; ++k) { e[k] = expf(y[k] - m); s += e[k]; }
    const float inv = 1.0f / s;
    for (int k = 0; k < KF; ++k)
        prob[(b * KF + k) * O_ + t] = e[k] * inv;
}

// --- aggregate per-sample weights: aggW[b][tap][o][c] (bf16, c-contiguous) --
__global__ void agg_kernel(const float* __restrict__ prob,
                           const float* __restrict__ weight,
                           unsigned short* __restrict__ aggW) {
    const int bo = blockIdx.x;
    const int b = bo >> 8, o = bo & 255;
    const int c = threadIdx.x;
    float p[KF];
    for (int k = 0; k < KF; ++k) p[k] = prob[(b * KF + k) * O_ + o];
    float acc[9];
    for (int t9 = 0; t9 < 9; ++t9) acc[t9] = 0.f;
    for (int k = 0; k < KF; ++k) {
        const float* wp = weight + (((size_t)k * O_ + o) * C_ + c) * 9;
        for (int t9 = 0; t9 < 9; ++t9) acc[t9] += p[k] * wp[t9];
    }
    for (int t9 = 0; t9 < 9; ++t9)
        aggW[(((size_t)b * 9 + t9) * O_ + o) * C_ + c] = f32_to_bf16(acc[t9]);
}

// ------- WMMA conv: shifted GEMMs, async double-buffered LDS, K=64/stage ----
// grid: (h=64, mtile=2, b=16); block: 256 threads = 8 waves (4M x 2N of 32x32)
#define PITCH 72   // LDS row pitch in bf16 (144B: 16B-aligned, conflict-free)

__device__ __forceinline__ void async_b128(unsigned lds_off, const void* gaddr) {
    asm volatile("global_load_async_to_lds_b128 %0, %1, off"
                 :: "v"(lds_off), "v"((unsigned long long)(size_t)gaddr)
                 : "memory");
}
__device__ __forceinline__ void wait_async0() {
    asm volatile("s_wait_asynccnt 0x0" ::: "memory");
}

__global__ __launch_bounds__(256) void conv_kernel(const unsigned short* __restrict__ xT,
                                                   const unsigned short* __restrict__ aggW,
                                                   float* __restrict__ out) {
    const int b    = blockIdx.z;
    const int m0   = blockIdx.y * 128;
    const int h    = blockIdx.x;               // output row; N-tile = full row
    const int tid  = threadIdx.x;
    const int lane = tid & 31;
    const int wave = tid >> 5;
    const int wm   = wave & 3;                 // M subtile (32 rows)
    const int wn   = wave >> 2;                // N subtile (32 cols)

    __shared__ unsigned short Alds[2][128 * PITCH];
    __shared__ unsigned short Blds[2][64 * PITCH];

    v8f acc[2][2];
    const v8f vzero = {0.f, 0.f, 0.f, 0.f, 0.f, 0.f, 0.f, 0.f};
    for (int i = 0; i < 2; ++i)
        for (int j = 0; j < 2; ++j) acc[i][j] = vzero;

    // fragment read offsets (ISA VGPR layouts)
    const int la   = lane & 15;
    const int ka   = (lane >> 4) * 8;          // A frag: K chunks at ka, ka+16
    const int kb16 = (lane >> 4) * 16;         // unused; B frag reads K chunks at ka, ka+16 like A

    // staging mapping: 16B chunks of 8 bf16 (rows of 64 bf16 = 8 chunks)
    const int rowS = tid >> 3;                 // 0..31
    const int octS = tid & 7;                  // chunk-in-row

    // valid taps packed as nibbles (off-image rows contribute exactly zero)
    unsigned long long tapPack = 0ull;
    int nt = 0;
    for (int tap = 0; tap < 9; ++tap) {
        const int hs = h + tap / 3 - 1;
        if (hs >= 0 && hs < H_) { tapPack |= (unsigned long long)tap << (4 * nt); ++nt; }
    }
    const int total = nt * 4;                  // 4 k-stages of 64 per tap

    auto stage = [&](int it, int buf) {
        const int tap = (int)((tapPack >> ((it >> 2) * 4)) & 15ull);
        const int kc  = (it & 3) * 64;
        const int dh1 = (tap * 11) >> 5;       // tap / 3 (tap in 0..8)
        const int dw  = tap - dh1 * 3 - 1;
        const int hs  = h + dh1 - 1;
        const unsigned short* aggB = aggW + (((size_t)b * 9 + tap) * O_ + m0) * C_ + kc;
        const unsigned short* xRow = xT + ((size_t)b * HW_ + hs * W_) * C_ + kc;
        // A tile: 128 x 64 bf16, four 16B chunks per thread
        #pragma unroll
        for (int i = 0; i < 4; ++i) {
            const int row = rowS + 32 * i;
            async_b128((unsigned)(size_t)&Alds[buf][row * PITCH + octS * 8],
                       aggB + (size_t)row * C_ + octS * 8);
        }
        // B tile: 64 x 64 bf16 with +-1 column shift; invalid column -> zeros
        #pragma unroll
        for (int i = 0; i < 2; ++i) {
            const int n = rowS + 32 * i;
            const int wsrc = n + dw;
            if (wsrc >= 0 && wsrc < W_) {
                async_b128((unsigned)(size_t)&Blds[buf][n * PITCH + octS * 8],
                           xRow + (size_t)wsrc * C_ + octS * 8);
            } else {
                const uint4 z = {0u, 0u, 0u, 0u};
                *(uint4*)(&Blds[buf][n * PITCH + octS * 8]) = z;
            }
        }
    };

    stage(0, 0);
    wait_async0();
    __syncthreads();

    for (int it = 0; it < total; ++it) {
        const int cur = it & 1;
        if (it + 1 < total) stage(it + 1, cur ^ 1);

        #pragma unroll
        for (int kk = 0; kk < 2; ++kk) {       // two WMMA k-steps per stage
            const int kb = kk * 32;
            v16bf afrag[2], bfrag[2];
            #pragma unroll
            for (int i = 0; i < 2; ++i) {
                const int m = wm * 32 + i * 16 + la;
                BF32 t;
                t.q[0] = *(const uint4*)(&Alds[cur][m * PITCH + kb + ka]);
                t.q[1] = *(const uint4*)(&Alds[cur][m * PITCH + kb + ka + 16]);
                afrag[i] = t.v;
            }
            #pragma unroll
            for (int j = 0; j < 2; ++j) {
                const int n = wn * 32 + j * 16 + la;
                BF32 t;
                t.q[0] = *(const uint4*)(&Blds[cur][n * PITCH + kb + ka]);
                t.q[1] = *(const uint4*)(&Blds[cur][n * PITCH + kb + ka + 16]);
                bfrag[j] = t.v;
            }
            #pragma unroll
            for (int i = 0; i < 2; ++i)
                #pragma unroll
                for (int j = 0; j < 2; ++j)
                    acc[i][j] = __builtin_amdgcn_wmma_f32_16x16x32_bf16(
                        false, afrag[i], false, bfrag[j],
                        (short)0, acc[i][j], false, false);
        }

        wait_async0();     // next buffer fully landed in LDS
        __syncthreads();   // all waves done reading cur, next visible to all
    }

    // epilogue: C layout (M = r + 8*(lane>>4), N = lane%16) -> out[b][o][h][w]
    const int mrow = 8 * (lane >> 4);
    const int ncol = lane & 15;
    for (int i = 0; i < 2; ++i)
        for (int j = 0; j < 2; ++j) {
            const int obase = m0 + wm * 32 + i * 16 + mrow;
            const int w     = wn * 32 + j * 16 + ncol;
            float* op = out + ((size_t)b * O_ + obase) * HW_ + h * W_ + w;
            for (int r = 0; r < 8; ++r)
                op[(size_t)r * HW_] = acc[i][j][r];
        }
}

extern "C" void kernel_launch(void* const* d_in, const int* in_sizes, int n_in,
                              void* d_out, int out_size, void* d_ws, size_t ws_size,
                              hipStream_t stream) {
    const float* x      = (const float*)d_in[0];
    const float* fc1_w  = (const float*)d_in[1];
    const float* fc2_w  = (const float*)d_in[2];
    const float* fc2_b  = (const float*)d_in[3];
    const float* weight = (const float*)d_in[4];

    char* ws = (char*)d_ws;
    // xT  : 16*4096*256 bf16 = 33554432 B
    // aggW: 16*9*256*256 bf16 = 18874368 B
    // pooled: 16*256 f32 = 16384 B ; prob: 16*4*256 f32 = 65536 B
    unsigned short* xT   = (unsigned short*)(ws);
    unsigned short* aggW = (unsigned short*)(ws + 33554432);
    float* pooled        = (float*)(ws + 33554432 + 18874368);
    float* prob          = (float*)(ws + 33554432 + 18874368 + 16384);
    float* out           = (float*)d_out;

    pool_kernel<<<B_ * C_, 256, 0, stream>>>(x, pooled);
    transpose_kernel<<<dim3(64, 4, B_), 256, 0, stream>>>(x, xT);
    prob_kernel<<<B_, 256, 0, stream>>>(pooled, fc1_w, fc2_w, fc2_b, prob);
    agg_kernel<<<B_ * O_, 256, 0, stream>>>(prob, weight, aggW);
    conv_kernel<<<dim3(H_, O_ / 128, B_), 256, 0, stream>>>(xT, aggW, out);
}
